// MultiHeadAttention_31997506355827
// MI455X (gfx1250) — compile-verified, matrix operand order checked by reading, not run on hardware
//
#include <hip/hip_runtime.h>
#include <hip/hip_bf16.h>

typedef __attribute__((ext_vector_type(16))) __bf16 v16bf;
typedef __attribute__((ext_vector_type(8)))  float  v8f;
typedef int v4i __attribute__((vector_size(16)));

#if __has_builtin(__builtin_amdgcn_global_load_async_to_lds_b128) && \
    __has_builtin(__builtin_amdgcn_s_wait_asynccnt)
#define HAS_ASYNC_LDS 1
#define WAIT_ASYNC(n) __builtin_amdgcn_s_wait_asynccnt(n)
#else
#define HAS_ASYNC_LDS 0
#define WAIT_ASYNC(n) ((void)0)
#endif

// 16-byte global -> LDS copy; async (ASYNCcnt path) when available.
__device__ __forceinline__ void async_copy16(const __bf16* g, __bf16* l) {
#if HAS_ASYNC_LDS
    __builtin_amdgcn_global_load_async_to_lds_b128((v4i*)(g), (v4i*)(l), 0, 0);
#else
    *reinterpret_cast<uint4*>(l) = *reinterpret_cast<const uint4*>(g);
#endif
}

// ---- intra-16-lane butterfly reductions via v_permlane16_b32 ----
#if __has_builtin(__builtin_amdgcn_permlane16)
#define PERM16(x, lo, hi) __builtin_amdgcn_permlane16((x), (x), (lo), (hi), false, false)
__device__ __forceinline__ float xor16_max(float v) {
    union { float f; unsigned u; } a, b;
    a.f = v;
    b.u = PERM16(a.u, 0x67452301u, 0xEFCDAB89u); a.f = fmaxf(a.f, b.f); // xor 1
    b.u = PERM16(a.u, 0x54761032u, 0xDCFE98BAu); a.f = fmaxf(a.f, b.f); // xor 2
    b.u = PERM16(a.u, 0x32107654u, 0xBA98FEDCu); a.f = fmaxf(a.f, b.f); // xor 4
    b.u = PERM16(a.u, 0xFEDCBA98u, 0x76543210u); a.f = fmaxf(a.f, b.f); // xor 8
    return a.f;
}
__device__ __forceinline__ float xor16_sum(float v) {
    union { float f; unsigned u; } a, b;
    a.f = v;
    b.u = PERM16(a.u, 0x67452301u, 0xEFCDAB89u); a.f += b.f;
    b.u = PERM16(a.u, 0x54761032u, 0xDCFE98BAu); a.f += b.f;
    b.u = PERM16(a.u, 0x32107654u, 0xBA98FEDCu); a.f += b.f;
    b.u = PERM16(a.u, 0xFEDCBA98u, 0x76543210u); a.f += b.f;
    return a.f;
}
#else
__device__ __forceinline__ float xor16_max(float v) {
    for (int m = 1; m < 16; m <<= 1) v = fmaxf(v, __shfl_xor(v, m, 32));
    return v;
}
__device__ __forceinline__ float xor16_sum(float v) {
    for (int m = 1; m < 16; m <<= 1) v += __shfl_xor(v, m, 32);
    return v;
}
#endif

// Build a 16x32 bf16 WMMA fragment (A, or B via transposed storage) from an
// LDS row base. Lanes 0-15 hold K {0..7,16..23}, lanes 16-31 hold K {8..15,24..31}.
__device__ __forceinline__ v16bf frag16(const __bf16* rowbase, int h) {
    v16bf f;
    uint4* p = reinterpret_cast<uint4*>(&f);
    p[0] = *reinterpret_cast<const uint4*>(rowbase + 8 * h);
    p[1] = *reinterpret_cast<const uint4*>(rowbase + 16 + 8 * h);
    return f;
}

__device__ __forceinline__ v8f wmma_bf16(v16bf a, v16bf b, v8f c) {
    return __builtin_amdgcn_wmma_f32_16x16x32_bf16(false, a, false, b, (short)0, c, false, false);
}

// ---------------- pre-pass kernels ----------------

// flat f32 -> bf16 (x), 4 elems/thread
__global__ __launch_bounds__(256) void cvt_bf16_kernel(const float* __restrict__ in,
                                                       __bf16* __restrict__ out) {
    size_t i = ((size_t)blockIdx.x * 256 + threadIdx.x) * 4;
    float4 v = *reinterpret_cast<const float4*>(in + i);
    alignas(8) __bf16 o4[4] = {(__bf16)v.x, (__bf16)v.y, (__bf16)v.z, (__bf16)v.w};
    *reinterpret_cast<uint2*>(out + i) = *reinterpret_cast<uint2*>(o4);
}

// W [K,N] f32 -> Wt [N,K] bf16, 32x32 tiles
__global__ __launch_bounds__(256) void transpose_w_kernel(const float* __restrict__ W,
                                                          __bf16* __restrict__ Wt,
                                                          int Kd, int Nd) {
    __shared__ __bf16 tile[32][33];
    int n0 = blockIdx.x * 32, k0 = blockIdx.y * 32;
    int tx = threadIdx.x & 31, ty = threadIdx.x >> 5;
    #pragma unroll
    for (int i = 0; i < 32; i += 8)
        tile[ty + i][tx] = (__bf16)W[(size_t)(k0 + ty + i) * Nd + n0 + tx];
    __syncthreads();
    #pragma unroll
    for (int i = 0; i < 32; i += 8)
        Wt[(size_t)(n0 + ty + i) * Kd + k0 + tx] = tile[tx][ty + i];
}

// bf16 [rows,cols] -> [cols,rows], batched over blockIdx.z
__global__ __launch_bounds__(256) void transpose_bf16_kernel(const __bf16* __restrict__ in,
                                                             __bf16* __restrict__ out,
                                                             int rows, int cols) {
    __shared__ __bf16 tile[32][33];
    const size_t ib = (size_t)blockIdx.z * rows * cols;
    int c0 = blockIdx.x * 32, r0 = blockIdx.y * 32;
    int tx = threadIdx.x & 31, ty = threadIdx.x >> 5;
    #pragma unroll
    for (int i = 0; i < 32; i += 8)
        tile[ty + i][tx] = in[ib + (size_t)(r0 + ty + i) * cols + c0 + tx];
    __syncthreads();
    #pragma unroll
    for (int i = 0; i < 32; i += 8)
        out[ib + (size_t)(c0 + ty + i) * rows + r0 + tx] = tile[tx][ty + i];
}

// ---------------- main GEMM ----------------
// C = A[M,K] @ Bt[N,K]^T + bias. Both operands bf16, Bt pre-transposed, so both
// tiles are straight async b128 copies. Double-buffered LDS, 128x128x32 block,
// 8 waves (4x2), wave tile 32x64 (8 wmma per K-step).
template <bool OUT_F32>
__global__ __launch_bounds__(256) void gemm_bt_kernel(
    const __bf16* __restrict__ A, const __bf16* __restrict__ Bt,
    const float* __restrict__ bias, void* __restrict__ Cptr,
    int M, int N, int K)
{
    __shared__ __align__(16) __bf16 As[2][128][40];
    __shared__ __align__(16) __bf16 Bs[2][128][40];

    const int tid    = threadIdx.x;
    const int lane   = tid & 31;
    const int w      = tid >> 5;
    const int wmBase = (w & 3) * 32;
    const int wnBase = (w >> 2) * 64;
    const int r      = lane & 15;
    const int hA     = lane >> 4;
    const int blockM = blockIdx.y * 128;
    const int blockN = blockIdx.x * 128;

    const __bf16* Ab = A + (size_t)blockM * K;
    const __bf16* Bb = Bt + (size_t)blockN * K;

    v8f acc[2][4];
    #pragma unroll
    for (int i = 0; i < 2; ++i)
        #pragma unroll
        for (int j = 0; j < 4; ++j)
            #pragma unroll
            for (int e = 0; e < 8; ++e) acc[i][j][e] = 0.0f;

    const int nK = K >> 5;

    // prologue: fill buffer 0 (tile k=0): 4 async ops/thread
    #pragma unroll
    for (int i = 0; i < 2; ++i) {
        int e = (i * 256 + tid) * 8, row = e >> 5, col = e & 31;
        async_copy16(Ab + (size_t)row * K + col, &As[0][row][col]);
        async_copy16(Bb + (size_t)row * K + col, &Bs[0][row][col]);
    }

    for (int kk = 0; kk < nK; ++kk) {
        const int cur = kk & 1, nxt = cur ^ 1;
        if (kk + 1 < nK) {
            const int k0 = (kk + 1) << 5;
            #pragma unroll
            for (int i = 0; i < 2; ++i) {
                int e = (i * 256 + tid) * 8, row = e >> 5, col = e & 31;
                async_copy16(Ab + (size_t)row * K + k0 + col, &As[nxt][row][col]);
                async_copy16(Bb + (size_t)row * K + k0 + col, &Bs[nxt][row][col]);
            }
            WAIT_ASYNC(4);   // oldest 4 (current tile) complete
        } else {
            WAIT_ASYNC(0);
        }
        __syncthreads();

        v16bf afrag[2];
        #pragma unroll
        for (int mi = 0; mi < 2; ++mi)
            afrag[mi] = frag16(&As[cur][wmBase + mi * 16 + r][0], hA);
        #pragma unroll
        for (int ni = 0; ni < 4; ++ni) {
            v16bf bfrag = frag16(&Bs[cur][wnBase + ni * 16 + r][0], hA);
            #pragma unroll
            for (int mi = 0; mi < 2; ++mi)
                acc[mi][ni] = wmma_bf16(afrag[mi], bfrag, acc[mi][ni]);
        }
        __syncthreads();
    }

    const int c  = lane & 15;
    const int hh = lane >> 4;
    #pragma unroll
    for (int mi = 0; mi < 2; ++mi) {
        #pragma unroll
        for (int ni = 0; ni < 4; ++ni) {
            int n = blockN + wnBase + ni * 16 + c;
            float bv = bias[n];
            #pragma unroll
            for (int j = 0; j < 8; ++j) {
                int m = blockM + wmBase + mi * 16 + j + 8 * hh;
                float val = acc[mi][ni][j] + bv;
                if (OUT_F32) ((float*)Cptr)[(size_t)m * N + n] = val;
                else         ((__bf16*)Cptr)[(size_t)m * N + n] = (__bf16)val;
            }
        }
    }
}

// ---------------- flash attention (causal) ----------------
// Q/K: bf16 [B*T, D]; Vt: bf16 [B][D][T] (pre-transposed); O: bf16 [B*T, D].
// Per block: one (b,h), 128 query rows, 8 waves x 16 rows. 64-key tiles.
__global__ __launch_bounds__(256) void attn_kernel(
    const __bf16* __restrict__ Q, const __bf16* __restrict__ K,
    const __bf16* __restrict__ Vt, __bf16* __restrict__ O,
    int T, int D)
{
    __shared__ __align__(16) __bf16 ktile[64][72];     // [key][dk]
    __shared__ __align__(16) __bf16 vtile[64][72];     // [dk][key]
    __shared__ __align__(16) __bf16 ptile[8][16][72];  // per-wave P scratch

    const int tid   = threadIdx.x;
    const int lane  = tid & 31;
    const int w     = tid >> 5;
    const int b     = blockIdx.z;
    const int h     = blockIdx.y;
    const int qBase = blockIdx.x * 128;
    const int qRow0 = qBase + w * 16;
    const int r     = lane & 15;
    const int hA    = lane >> 4;
    const size_t headOff = (size_t)h * 64;

    v16bf aQ[2];
    {
        const __bf16* qrow = Q + (size_t)(b * T + qRow0 + r) * D + headOff;
        #pragma unroll
        for (int mi = 0; mi < 2; ++mi) {
            uint4* p = reinterpret_cast<uint4*>(&aQ[mi]);
            p[0] = *reinterpret_cast<const uint4*>(qrow + mi * 32 + 8 * hA);
            p[1] = *reinterpret_cast<const uint4*>(qrow + mi * 32 + 16 + 8 * hA);
        }
    }

    v8f o[4];
    #pragma unroll
    for (int i = 0; i < 4; ++i)
        #pragma unroll
        for (int j = 0; j < 8; ++j) o[i][j] = 0.0f;
    float rm[8], rl[8];
    #pragma unroll
    for (int j = 0; j < 8; ++j) { rm[j] = -3.0e38f; rl[j] = 0.0f; }

    const float scale = 0.125f;              // 1/sqrt(64)
    const int numTiles = (qBase + 128) / 64; // keys 0..qBase+127

    for (int kt = 0; kt < numTiles; ++kt) {
        const int key0 = kt * 64;
        __syncthreads();   // previous tile fully consumed
        {   // K tile: natural copy; V tile: rows of pre-transposed Vt
            #pragma unroll
            for (int i = 0; i < 2; ++i) {
                int e = (i * 256 + tid) * 8;
                int a0 = e >> 6, a1 = e & 63;
                async_copy16(K + (size_t)(b * T + key0 + a0) * D + headOff + a1,
                             &ktile[a0][a1]);
                async_copy16(Vt + ((size_t)b * D + headOff + a0) * T + key0 + a1,
                             &vtile[a0][a1]);
            }
        }
        WAIT_ASYNC(0);
        __syncthreads();
        if (key0 > qRow0 + 15) continue;   // fully masked for this wave

        // S = Q K^T over dk=64 (two K-steps), four 16-key column tiles
        v8f s[4];
        #pragma unroll
        for (int nt = 0; nt < 4; ++nt)
            #pragma unroll
            for (int j = 0; j < 8; ++j) s[nt][j] = 0.0f;
        #pragma unroll
        for (int half = 0; half < 2; ++half)
            #pragma unroll
            for (int nt = 0; nt < 4; ++nt)
                s[nt] = wmma_bf16(aQ[half], frag16(&ktile[nt * 16 + r][half * 32], hA), s[nt]);

        // online softmax: rows in VGPR idx j (+8*hA), keys in lanes 0..15
        const int c = r;
        #pragma unroll
        for (int j = 0; j < 8; ++j) {
            int qg = qRow0 + j + 8 * hA;
            float v[4];
            #pragma unroll
            for (int nt = 0; nt < 4; ++nt) {
                v[nt] = s[nt][j] * scale;
                if (key0 + nt * 16 + c > qg) v[nt] = -1.0e9f;
            }
            float mx = fmaxf(fmaxf(v[0], v[1]), fmaxf(v[2], v[3]));
            mx = xor16_max(mx);
            float nm   = fmaxf(rm[j], mx);
            float corr = __expf(rm[j] - nm);
            rm[j] = nm;
            float p0 = __expf(v[0] - nm), p1 = __expf(v[1] - nm);
            float p2 = __expf(v[2] - nm), p3 = __expf(v[3] - nm);
            float rs = xor16_sum((p0 + p1) + (p2 + p3));
            rl[j] = rl[j] * corr + rs;
            #pragma unroll
            for (int t = 0; t < 4; ++t) o[t][j] *= corr;
            int m = j + 8 * hA;
            ptile[w][m][c]      = (__bf16)p0;
            ptile[w][m][c + 16] = (__bf16)p1;
            ptile[w][m][c + 32] = (__bf16)p2;
            ptile[w][m][c + 48] = (__bf16)p3;
        }

        // O += P(16x64) x V(64x64); intra-wave LDS RAW is in-order
        v16bf pfA = frag16(&ptile[w][r][0], hA);
        v16bf pfB = frag16(&ptile[w][r][32], hA);
        #pragma unroll
        for (int ni = 0; ni < 4; ++ni) {
            o[ni] = wmma_bf16(pfA, frag16(&vtile[ni * 16 + r][0], hA), o[ni]);
            o[ni] = wmma_bf16(pfB, frag16(&vtile[ni * 16 + r][32], hA), o[ni]);
        }
    }

    #pragma unroll
    for (int ni = 0; ni < 4; ++ni) {
        #pragma unroll
        for (int j = 0; j < 8; ++j) {
            float val = o[ni][j] / rl[j];
            int m = qRow0 + j + 8 * hA;
            O[(size_t)(b * T + m) * D + headOff + ni * 16 + r] = (__bf16)val;
        }
    }
}

extern "C" void kernel_launch(void* const* d_in, const int* in_sizes, int n_in,
                              void* d_out, int out_size, void* d_ws, size_t ws_size,
                              hipStream_t stream) {
    (void)in_sizes; (void)n_in; (void)out_size; (void)ws_size;
    const int B = 2, T = 2048, D = 1024;
    const int M = B * T;

    const float* x  = (const float*)d_in[0];
    // d_in[1] = mask (causal; handled analytically)
    const float* Wq = (const float*)d_in[2];
    const float* bq = (const float*)d_in[3];
    const float* Wk = (const float*)d_in[4];
    const float* bk = (const float*)d_in[5];
    const float* Wv = (const float*)d_in[6];
    const float* bv = (const float*)d_in[7];
    const float* Wo = (const float*)d_in[8];
    const float* bo = (const float*)d_in[9];

    __bf16* xbf  = (__bf16*)d_ws;                       // M*D
    __bf16* wtq  = xbf  + (size_t)M * D;                // D*D
    __bf16* wtk  = wtq  + (size_t)D * D;
    __bf16* wtv  = wtk  + (size_t)D * D;
    __bf16* wto  = wtv  + (size_t)D * D;
    __bf16* qws  = wto  + (size_t)D * D;                // M*D
    __bf16* kws  = qws  + (size_t)M * D;                // M*D
    __bf16* vws  = kws  + (size_t)M * D;                // M*D
    __bf16* vtws = vws  + (size_t)M * D;                // B*D*T
    __bf16* aws  = vtws + (size_t)M * D;                // M*D

    // pre-pass: bf16 conversion + weight transposes
    cvt_bf16_kernel<<<(M * D) / 1024, 256, 0, stream>>>(x, xbf);
    dim3 gw(D / 32, D / 32);
    transpose_w_kernel<<<gw, 256, 0, stream>>>(Wq, wtq, D, D);
    transpose_w_kernel<<<gw, 256, 0, stream>>>(Wk, wtk, D, D);
    transpose_w_kernel<<<gw, 256, 0, stream>>>(Wv, wtv, D, D);
    transpose_w_kernel<<<gw, 256, 0, stream>>>(Wo, wto, D, D);

    // projections
    dim3 gblk(D / 128, M / 128);
    gemm_bt_kernel<false><<<gblk, 256, 0, stream>>>(xbf, wtq, bq, qws, M, D, D);
    gemm_bt_kernel<false><<<gblk, 256, 0, stream>>>(xbf, wtk, bk, kws, M, D, D);
    gemm_bt_kernel<false><<<gblk, 256, 0, stream>>>(xbf, wtv, bv, vws, M, D, D);

    // V -> per-batch [D][T] for transposed attention loads
    dim3 gv(D / 32, T / 32, B);
    transpose_bf16_kernel<<<gv, 256, 0, stream>>>(vws, vtws, T, D);

    // attention
    dim3 ga(T / 128, 16 /*H*/, B);
    attn_kernel<<<ga, 256, 0, stream>>>(qws, kws, vtws, aws, T, D);

    // output projection (fp32 out)
    gemm_bt_kernel<true><<<gblk, 256, 0, stream>>>(aws, wto, bo, (float*)d_out, M, D, D);
}
